// Atten_86552180949544
// MI455X (gfx1250) — matrix operand order repaired and checked by reading, not compiled, not run on hardware
//
#include <hip/hip_runtime.h>
#include <hip/hip_bf16.h>

typedef __bf16 bf16;
typedef __attribute__((ext_vector_type(16))) __bf16 v16bf;
typedef __attribute__((ext_vector_type(8)))  __bf16 v8bf;
typedef __attribute__((ext_vector_type(8)))  float  v8f;

#define D_   768
#define L_   2048
#define H_   12
#define HD_  64
#define NTOK 8192   // B*L

// ---------------------------------------------------------------------------
// WMMA helper (CDNA5 wave32, 16x16x32 bf16 -> f32)
// ---------------------------------------------------------------------------
__device__ __forceinline__ v8f wmma_bf16(v16bf a, v16bf b, v8f c) {
    return __builtin_amdgcn_wmma_f32_16x16x32_bf16(
        false, a, false, b, (short)0, c, false, false);
}

// A matrix 16x32 (MxK), bf16 row-major source, leading dim ld.
__device__ __forceinline__ v16bf load_a_frag(const bf16* base, int ld, int k0,
                                             int ln, int lh) {
    const bf16* p = base + (size_t)ln * ld + k0 + lh * 16;
    v8bf lo = *(const v8bf*)(p);
    v8bf hi = *(const v8bf*)(p + 8);
    v16bf r;
#pragma unroll
    for (int i = 0; i < 8; ++i) { r[i] = lo[i]; r[i + 8] = hi[i]; }
    return r;
}

// B matrix 32x16 (KxN), stored N-major with K contiguous (B^T row-major).
__device__ __forceinline__ v16bf load_b_frag(const bf16* base, int ld, int k0,
                                             int ln, int lh) {
    return *(const v16bf*)(base + (size_t)ln * ld + k0 + lh * 16);
}

// ---------------------------------------------------------------------------
// CDNA5 async global -> LDS copy (16 bytes per lane, tracked by ASYNCcnt)
// ---------------------------------------------------------------------------
__device__ __forceinline__ void async_b128(unsigned int lds_off,
                                           const void* gaddr) {
    asm volatile("global_load_async_to_lds_b128 %0, %1, off"
                 :
                 : "v"(lds_off), "v"((unsigned long long)(uintptr_t)gaddr)
                 : "memory");
}
__device__ __forceinline__ void wait_async0() {
    asm volatile("s_wait_asynccnt 0x0" ::: "memory");
}

// ---------------------------------------------------------------------------
// fp32 -> bf16 conversion (8 elems/thread, b128-aligned store)
// ---------------------------------------------------------------------------
__global__ void cvt_f32_bf16(const float* __restrict__ s, bf16* __restrict__ d,
                             int n) {
    int i = (blockIdx.x * blockDim.x + threadIdx.x) * 8;
    if (i + 7 < n) {
        float4 a = *(const float4*)(s + i);
        float4 b = *(const float4*)(s + i + 4);
        v8bf r;
        r[0] = (bf16)a.x; r[1] = (bf16)a.y; r[2] = (bf16)a.z; r[3] = (bf16)a.w;
        r[4] = (bf16)b.x; r[5] = (bf16)b.y; r[6] = (bf16)b.z; r[7] = (bf16)b.w;
        *(v8bf*)(d + i) = r;
    }
}

// ---------------------------------------------------------------------------
// Fused QKV projection: y = x @ W^T (+bias).  grid=(32,12,3), block=256.
// Wave tile = 32 rows x 64 outputs; K-loop ping-pong pipelined (no reg copies).
// ---------------------------------------------------------------------------
__global__ __launch_bounds__(256) void qkv_proj(
    const bf16* __restrict__ xb,
    const bf16* __restrict__ wq, const bf16* __restrict__ wk,
    const bf16* __restrict__ wv,
    const float* __restrict__ bq, const float* __restrict__ bv,
    bf16* __restrict__ qb, bf16* __restrict__ kb, bf16* __restrict__ vtb) {
    const int wid = threadIdx.x >> 5;
    const int lane = threadIdx.x & 31;
    const int ln = lane & 15, lh = lane >> 4;
    const int m0 = blockIdx.x * 256 + wid * 32;
    const int head = blockIdx.y;
    const int o0 = head * HD_;
    const int z = blockIdx.z;
    const bf16* W = (z == 0) ? wq : (z == 1) ? wk : wv;

    const bf16* arow0 = xb + (size_t)m0 * D_;
    const bf16* arow1 = arow0 + (size_t)16 * D_;
    const bf16* brow[4];
#pragma unroll
    for (int dt = 0; dt < 4; ++dt) brow[dt] = W + (size_t)(o0 + dt * 16) * D_;

    auto load_set = [&](v16bf& A0, v16bf& A1, v16bf* B, int kk) {
        A0 = load_a_frag(arow0, D_, kk, ln, lh);
        A1 = load_a_frag(arow1, D_, kk, ln, lh);
#pragma unroll
        for (int dt = 0; dt < 4; ++dt)
            B[dt] = load_b_frag(brow[dt], D_, kk, ln, lh);
    };

    v8f vz = {};
    v8f acc0[4] = {vz, vz, vz, vz};
    v8f acc1[4] = {vz, vz, vz, vz};

    v16bf A0a, A1a, Ba[4];      // ping set (kk)
    v16bf A0b, A1b, Bb[4];      // pong set (kk + 32)
    load_set(A0a, A1a, Ba, 0);
    load_set(A0b, A1b, Bb, 32);

#pragma unroll 1
    for (int kk = 0; kk < D_; kk += 64) {
#pragma unroll
        for (int dt = 0; dt < 4; ++dt) {
            acc0[dt] = wmma_bf16(A0a, Ba[dt], acc0[dt]);
            acc1[dt] = wmma_bf16(A1a, Ba[dt], acc1[dt]);
        }
        if (kk + 64 < D_) load_set(A0a, A1a, Ba, kk + 64);
#pragma unroll
        for (int dt = 0; dt < 4; ++dt) {
            acc0[dt] = wmma_bf16(A0b, Bb[dt], acc0[dt]);
            acc1[dt] = wmma_bf16(A1b, Bb[dt], acc1[dt]);
        }
        if (kk + 96 < D_) load_set(A0b, A1b, Bb, kk + 96);
    }

#pragma unroll
    for (int r = 0; r < 2; ++r) {
        v8f* acc = r ? acc1 : acc0;
#pragma unroll
        for (int dt = 0; dt < 4; ++dt)
#pragma unroll
            for (int j = 0; j < 8; ++j) {
                float v = acc[dt][j];
                int o = o0 + dt * 16 + ln;
                int token = m0 + r * 16 + j + 8 * lh;
                int b = token >> 11, l = token & (L_ - 1);
                int d = dt * 16 + ln;
                size_t bh = (size_t)b * H_ + head;
                if (z == 0) {
                    v = (v + bq[o]) * 0.125f;
                    qb[(bh * L_ + l) * HD_ + d] = (bf16)v;
                } else if (z == 1) {
                    kb[(bh * L_ + l) * HD_ + d] = (bf16)v;
                } else {
                    v += bv[o];
                    vtb[(bh * HD_ + d) * L_ + l] = (bf16)v;
                }
            }
    }
}

// ---------------------------------------------------------------------------
// Flash attention with double-buffered async K/V staging.
// grid=(16,48), block=256 (8 waves x 16 q-rows).
// ---------------------------------------------------------------------------
__global__ __launch_bounds__(256) void flash_attn(
    const bf16* __restrict__ qbuf, const bf16* __restrict__ kbuf,
    const bf16* __restrict__ vtbuf, const float* __restrict__ mask,
    bf16* __restrict__ abuf) {
    __shared__ bf16 sK[2][64 * 64];       // [key][d]
    __shared__ bf16 sV[2][64 * 64];       // [d][key]
    __shared__ bf16 sP[8][16 * 64];       // per-wave P staging

    const int wid = threadIdx.x >> 5;
    const int lane = threadIdx.x & 31;
    const int ln = lane & 15, lh = lane >> 4;

    const int bh = blockIdx.y;
    const int b = bh / H_, h = bh % H_;
    const bf16* Q  = qbuf  + (size_t)bh * L_ * HD_;
    const bf16* K  = kbuf  + (size_t)bh * L_ * HD_;
    const bf16* Vt = vtbuf + (size_t)bh * L_ * HD_;
    const float* M = mask  + (size_t)b * L_ * L_;
    const int q0 = blockIdx.x * 128 + wid * 16;

    v16bf aq0 = load_a_frag(Q + (size_t)q0 * HD_, HD_, 0,  ln, lh);
    v16bf aq1 = load_a_frag(Q + (size_t)q0 * HD_, HD_, 32, ln, lh);

    v8f vz = {};
    v8f o[4] = {vz, vz, vz, vz};
    float mrow[8], lrow[8];
#pragma unroll
    for (int j = 0; j < 8; ++j) { mrow[j] = -INFINITY; lrow[j] = 0.0f; }

    // cooperative async stage of one 64-key block into buffer `buf`
    auto issue = [&](int buf, int kbi) {
#pragma unroll
        for (int c = 0; c < 2; ++c) {
            int idx = c * 256 + threadIdx.x;       // 512 b128 chunks per tile
            int row = idx >> 3;                    // 0..63
            int col = (idx & 7) * 8;               // bf16 elems, 16B chunks
            const bf16* gk = K + (size_t)(kbi * 64 + row) * HD_ + col;
            const bf16* gv = Vt + (size_t)row * L_ + kbi * 64 + col;
            async_b128((unsigned int)(uintptr_t)&sK[buf][row * 64 + col], gk);
            async_b128((unsigned int)(uintptr_t)&sV[buf][row * 64 + col], gv);
        }
    };

    issue(0, 0);

    for (int kbi = 0; kbi < L_ / 64; ++kbi) {
        const int cur = kbi & 1;
        wait_async0();        // own async writes for buf[cur] complete
        __syncthreads();      // all waves' writes visible; prev compute done
        if (kbi + 1 < L_ / 64) issue(1 - cur, kbi + 1);

        const bf16* sKb = &sK[cur][0];
        const bf16* sVb = &sV[cur][0];

        // S = Q K^T (+mask)
        v8f s[4];
#pragma unroll
        for (int nt = 0; nt < 4; ++nt) {
            v8f t = vz;
            v16bf bk0 = load_b_frag(sKb + nt * 16 * 64, 64, 0,  ln, lh);
            v16bf bk1 = load_b_frag(sKb + nt * 16 * 64, 64, 32, ln, lh);
            t = wmma_bf16(aq0, bk0, t);
            t = wmma_bf16(aq1, bk1, t);
#pragma unroll
            for (int j = 0; j < 8; ++j)
                t[j] += M[(size_t)(q0 + j + 8 * lh) * L_ + kbi * 64 + nt * 16 + ln];
            s[nt] = t;
        }

        // online softmax
#pragma unroll
        for (int j = 0; j < 8; ++j) {
            float v = fmaxf(fmaxf(s[0][j], s[1][j]), fmaxf(s[2][j], s[3][j]));
            v = fmaxf(v, __shfl_xor(v, 1, 32));
            v = fmaxf(v, __shfl_xor(v, 2, 32));
            v = fmaxf(v, __shfl_xor(v, 4, 32));
            v = fmaxf(v, __shfl_xor(v, 8, 32));
            float mn = fmaxf(mrow[j], v);
            float sc = __expf(mrow[j] - mn);
            mrow[j] = mn;
            float rs = 0.0f;
#pragma unroll
            for (int nt = 0; nt < 4; ++nt) {
                float p = __expf(s[nt][j] - mn);
                s[nt][j] = p;
                rs += p;
            }
            rs += __shfl_xor(rs, 1, 32);
            rs += __shfl_xor(rs, 2, 32);
            rs += __shfl_xor(rs, 4, 32);
            rs += __shfl_xor(rs, 8, 32);
            lrow[j] = lrow[j] * sc + rs;
#pragma unroll
            for (int dt = 0; dt < 4; ++dt) o[dt][j] *= sc;
        }

        // P: C-layout -> LDS -> A-layout (per-wave, in-order DS)
        bf16* P = sP[wid];
#pragma unroll
        for (int nt = 0; nt < 4; ++nt)
#pragma unroll
            for (int j = 0; j < 8; ++j)
                P[(j + 8 * lh) * 64 + nt * 16 + ln] = (bf16)s[nt][j];

        v16bf ap0 = load_a_frag(P, 64, 0,  ln, lh);
        v16bf ap1 = load_a_frag(P, 64, 32, ln, lh);
#pragma unroll
        for (int dt = 0; dt < 4; ++dt) {
            v16bf bv0 = load_b_frag(sVb + dt * 16 * 64, 64, 0,  ln, lh);
            v16bf bv1 = load_b_frag(sVb + dt * 16 * 64, 64, 32, ln, lh);
            o[dt] = wmma_bf16(ap0, bv0, o[dt]);
            o[dt] = wmma_bf16(ap1, bv1, o[dt]);
        }
    }

    float inv[8];
#pragma unroll
    for (int j = 0; j < 8; ++j) inv[j] = 1.0f / lrow[j];
#pragma unroll
    for (int dt = 0; dt < 4; ++dt)
#pragma unroll
        for (int j = 0; j < 8; ++j) {
            int qrow = q0 + j + 8 * lh;
            size_t token = (size_t)b * L_ + qrow;
            int feat = h * HD_ + dt * 16 + ln;
            abuf[token * D_ + feat] = (bf16)(o[dt][j] * inv[j]);
        }
}

// ---------------------------------------------------------------------------
// Output projection: out = attn @ Wo^T + bo (fp32).  grid=(32,12), block=256.
// ---------------------------------------------------------------------------
__global__ __launch_bounds__(256) void out_proj(
    const bf16* __restrict__ ab, const bf16* __restrict__ wo,
    const float* __restrict__ bo, float* __restrict__ out) {
    const int wid = threadIdx.x >> 5;
    const int lane = threadIdx.x & 31;
    const int ln = lane & 15, lh = lane >> 4;
    const int m0 = blockIdx.x * 256 + wid * 32;
    const int o0 = blockIdx.y * 64;

    const bf16* arow0 = ab + (size_t)m0 * D_;
    const bf16* arow1 = arow0 + (size_t)16 * D_;
    const bf16* brow[4];
#pragma unroll
    for (int dt = 0; dt < 4; ++dt) brow[dt] = wo + (size_t)(o0 + dt * 16) * D_;

    auto load_set = [&](v16bf& A0, v16bf& A1, v16bf* B, int kk) {
        A0 = load_a_frag(arow0, D_, kk, ln, lh);
        A1 = load_a_frag(arow1, D_, kk, ln, lh);
#pragma unroll
        for (int dt = 0; dt < 4; ++dt)
            B[dt] = load_b_frag(brow[dt], D_, kk, ln, lh);
    };

    v8f vz = {};
    v8f acc0[4] = {vz, vz, vz, vz};
    v8f acc1[4] = {vz, vz, vz, vz};

    v16bf A0a, A1a, Ba[4];
    v16bf A0b, A1b, Bb[4];
    load_set(A0a, A1a, Ba, 0);
    load_set(A0b, A1b, Bb, 32);

#pragma unroll 1
    for (int kk = 0; kk < D_; kk += 64) {
#pragma unroll
        for (int dt = 0; dt < 4; ++dt) {
            acc0[dt] = wmma_bf16(A0a, Ba[dt], acc0[dt]);
            acc1[dt] = wmma_bf16(A1a, Ba[dt], acc1[dt]);
        }
        if (kk + 64 < D_) load_set(A0a, A1a, Ba, kk + 64);
#pragma unroll
        for (int dt = 0; dt < 4; ++dt) {
            acc0[dt] = wmma_bf16(A0b, Bb[dt], acc0[dt]);
            acc1[dt] = wmma_bf16(A1b, Bb[dt], acc1[dt]);
        }
        if (kk + 96 < D_) load_set(A0b, A1b, Bb, kk + 96);
    }

#pragma unroll
    for (int r = 0; r < 2; ++r) {
        v8f* acc = r ? acc1 : acc0;
#pragma unroll
        for (int dt = 0; dt < 4; ++dt)
#pragma unroll
            for (int j = 0; j < 8; ++j) {
                int o = o0 + dt * 16 + ln;
                int token = m0 + r * 16 + j + 8 * lh;
                out[(size_t)token * D_ + o] = acc[dt][j] + bo[o];
            }
    }
}

// ---------------------------------------------------------------------------
extern "C" void kernel_launch(void* const* d_in, const int* in_sizes, int n_in,
                              void* d_out, int out_size, void* d_ws,
                              size_t ws_size, hipStream_t stream) {
    const float* x    = (const float*)d_in[0];
    const float* mask = (const float*)d_in[1];
    const float* Wq   = (const float*)d_in[2];
    const float* bq   = (const float*)d_in[3];
    const float* Wk   = (const float*)d_in[4];
    const float* Wv   = (const float*)d_in[5];
    const float* bv   = (const float*)d_in[6];
    const float* Wo   = (const float*)d_in[7];
    const float* bo   = (const float*)d_in[8];
    float* out = (float*)d_out;

    const size_t NX = (size_t)NTOK * D_;
    const size_t NW = (size_t)D_ * D_;

    bf16* xb  = (bf16*)d_ws;
    bf16* wqb = xb + NX;
    bf16* wkb = wqb + NW;
    bf16* wvb = wkb + NW;
    bf16* wob = wvb + NW;
    bf16* qb  = wob + NW;
    bf16* kb  = qb + NX;
    bf16* vtb = kb + NX;
    bf16* ab  = vtb + NX;

    cvt_f32_bf16<<<dim3((unsigned)(NX / 8 / 256)), 256, 0, stream>>>(x, xb, (int)NX);
    cvt_f32_bf16<<<dim3((unsigned)(NW / 8 / 256)), 256, 0, stream>>>(Wq, wqb, (int)NW);
    cvt_f32_bf16<<<dim3((unsigned)(NW / 8 / 256)), 256, 0, stream>>>(Wk, wkb, (int)NW);
    cvt_f32_bf16<<<dim3((unsigned)(NW / 8 / 256)), 256, 0, stream>>>(Wv, wvb, (int)NW);
    cvt_f32_bf16<<<dim3((unsigned)(NW / 8 / 256)), 256, 0, stream>>>(Wo, wob, (int)NW);

    qkv_proj<<<dim3(32, 12, 3), 256, 0, stream>>>(xb, wqb, wkb, wvb, bq, bv,
                                                  qb, kb, vtb);
    flash_attn<<<dim3(16, 48), 256, 0, stream>>>(qb, kb, vtb, mask, ab);
    out_proj<<<dim3(32, 12), 256, 0, stream>>>(ab, wob, bo, out);
}
